// GlobalMamba_40346922778710
// MI455X (gfx1250) — compile-verified, matrix operand order checked by reading, not run on hardware
//
#include <hip/hip_runtime.h>
#include <hip/hip_bf16.h>

// ---------------------------------------------------------------------------
// Mamba block forward for MI455X (gfx1250, wave32).
// GEMMs: v_wmma_f32_16x16x32_bf16, 32x64 tile per wave, shared B tile staged
// into LDS with global_load_async_to_lds_b128 (double buffered, ASYNCcnt).
// All 4 B fragments live in distinct registers so the 8 WMMAs per K-step can
// issue back-to-back after a single s_wait.
// Scan: per-(b,d,n) lanes with shfl_xor reduction across the 16 states.
// ---------------------------------------------------------------------------

#define D_MODEL 512
#define D_STATE 16
#define D_CONV 4
#define D_INNER 1024
#define DT_RANK 32
#define BATCH 2
#define SEQLEN 2048
#define ROWS (BATCH * SEQLEN)   // 4096

typedef __attribute__((ext_vector_type(16))) __bf16 v16bf;
typedef __attribute__((ext_vector_type(8)))  __bf16 v8bf;
typedef __attribute__((ext_vector_type(8)))  float  v8f;

__device__ __forceinline__ unsigned short f32_to_bf16_rne(float f) {
  unsigned int u = __float_as_uint(f);
  unsigned int r = u + 0x7FFFu + ((u >> 16) & 1u);   // round-to-nearest-even
  return (unsigned short)(r >> 16);
}

// ---- CDNA5 async copy: 16 bytes global -> LDS, tracked by ASYNCcnt --------
__device__ __forceinline__ void async_copy_b128(unsigned lds_off, const void* gaddr) {
  asm volatile("global_load_async_to_lds_b128 %0, %1, off"
               :: "v"(lds_off), "v"(gaddr)
               : "memory");
}
__device__ __forceinline__ void wait_asynccnt_le2() {
  asm volatile("s_wait_asynccnt 0x2" ::: "memory");
}
__device__ __forceinline__ void wait_asynccnt_0() {
  asm volatile("s_wait_asynccnt 0x0" ::: "memory");
}

// ---------------- f32 -> bf16 bulk convert ----------------
__global__ void cvt_bf16_kernel(const float* __restrict__ in,
                                unsigned short* __restrict__ out, int n) {
  int i = blockIdx.x * blockDim.x + threadIdx.x;
  if (i < n) out[i] = f32_to_bf16_rne(in[i]);
}

// ---------------- WMMA GEMM: C[M,N] = A[M,K] * W[N,K]^T ----------------
// A, W row-major bf16; C row-major f32. M%128==0, N%64==0, K%32==0.
// Block = 128 threads = 4 waves stacked along M (block tile 128M x 64N).
// Each wave: 32M x 64N = 2 A-frags x 4 B-frags = 8 WMMAs per K-step.
__global__ void wmma_gemm_bf16_kernel(const __bf16* __restrict__ A,
                                      const __bf16* __restrict__ W,
                                      float* __restrict__ C,
                                      int M, int N, int K) {
  __shared__ __align__(16) unsigned char Bs[2][4096];   // 2 x (64 rows x 64B)

  const int wave = threadIdx.x >> 5;
  const int lane = threadIdx.x & 31;
  const int m0 = blockIdx.y * 128 + wave * 32;
  const int n0 = blockIdx.x * 64;

  const int lm = lane & 15;     // row within 16x16 tile
  const int hi = lane >> 4;     // K-half select (ISA 16-bit A-fragment layout)

  // A fragment pointers for the 2 m-tiles of this wave
  const __bf16* arow0 = A + (size_t)(m0 + lm) * K + hi * 8;
  const __bf16* arow1 = arow0 + (size_t)16 * K;

  // Async staging layout: LDS image of B tile is row-major, 64 rows x 64B.
  // Thread t copies bytes [t*16, t*16+16) and the same column 32 rows below.
  const int nl = threadIdx.x >> 2;            // 0..31  (row of first chunk)
  const int kb = (threadIdx.x & 3) * 16;      // byte column within 64B row
  const char* gB0 = (const char*)(W + (size_t)(n0 + nl) * K) + kb;       // row nl
  const char* gB1 = gB0 + (size_t)32 * K * 2;                            // row nl+32
  const unsigned ldsBase = (unsigned)(unsigned long long)(void*)&Bs[0][0];
  const unsigned ldsOffA = (unsigned)threadIdx.x * 16u;                  // chunk 0
  const unsigned ldsOffB = ldsOffA + 2048u;                              // chunk 1

  v8f acc[2][4];
#pragma unroll
  for (int i = 0; i < 2; ++i)
#pragma unroll
    for (int j = 0; j < 4; ++j) acc[i][j] = {};

  const int nk = K / 32;

  // Prologue: stage K-tile 0 into buffer 0.
  async_copy_b128(ldsBase + ldsOffA, gB0);
  async_copy_b128(ldsBase + ldsOffB, gB1);

  for (int kt = 0; kt < nk; ++kt) {
    const int buf = kt & 1;
    const int k0 = kt * 32;

    if (kt + 1 < nk) {   // stage next tile into the other buffer
      unsigned nb = ldsBase + (unsigned)((buf ^ 1) * 4096);
      size_t gofs = (size_t)(k0 + 32) * 2;   // 32 bf16 = 64B per K-step
      async_copy_b128(nb + ldsOffA, gB0 + gofs);
      async_copy_b128(nb + ldsOffB, gB1 + gofs);
      wait_asynccnt_le2();   // oldest 2 (current buffer) have landed
    } else {
      wait_asynccnt_0();
    }
    __syncthreads();         // all waves' portions of current tile visible

    // A fragments (global, b128 x2 each), with next-tile prefetch
    __builtin_prefetch(arow0 + k0 + 64, 0, 1);
    __builtin_prefetch(arow1 + k0 + 64, 0, 1);
    v8bf a_lo0 = *(const v8bf*)(arow0 + k0);
    v8bf a_hi0 = *(const v8bf*)(arow0 + k0 + 16);
    v8bf a_lo1 = *(const v8bf*)(arow1 + k0);
    v8bf a_hi1 = *(const v8bf*)(arow1 + k0 + 16);
    v16bf af[2];
#pragma unroll
    for (int i = 0; i < 8; ++i) {
      af[0][i] = a_lo0[i]; af[0][8 + i] = a_hi0[i];
      af[1][i] = a_lo1[i]; af[1][8 + i] = a_hi1[i];
    }

    // Load ALL 4 B fragments into distinct registers first (8x ds_load_b128,
    // one wait), so the 8 WMMAs below issue back-to-back with no LDS stalls.
    const unsigned char* bs = &Bs[buf][0];
    v16bf bfv[4];
#pragma unroll
    for (int j = 0; j < 4; ++j) {
      const unsigned char* brow = bs + (unsigned)(16 * j + lm) * 64 + hi * 16;
      v8bf b_lo = *(const v8bf*)(brow);
      v8bf b_hi = *(const v8bf*)(brow + 32);
#pragma unroll
      for (int i = 0; i < 8; ++i) { bfv[j][i] = b_lo[i]; bfv[j][8 + i] = b_hi[i]; }
    }

#pragma unroll
    for (int j = 0; j < 4; ++j)
#pragma unroll
      for (int i = 0; i < 2; ++i)
        acc[i][j] = __builtin_amdgcn_wmma_f32_16x16x32_bf16(
            false, af[i], false, bfv[j], (short)0, acc[i][j], false, false);

    __syncthreads();         // protect buf from next iteration's overwrite
  }

  // C/D layout: element e -> row (tile_m + e + 8*hi), col (tile_n + lm)
#pragma unroll
  for (int i = 0; i < 2; ++i) {
    float* cbase = C + (size_t)(m0 + 16 * i + hi * 8) * N + lm;
#pragma unroll
    for (int j = 0; j < 4; ++j) {
      float* crow = cbase + n0 + 16 * j;
#pragma unroll
      for (int e = 0; e < 8; ++e) crow[(size_t)e * N] = acc[i][j][e];
    }
  }
}

// ---------------- causal depthwise conv1d + SiLU ----------------
__global__ void conv_silu_kernel(const float* __restrict__ xz,
                                 const float* __restrict__ conv_w,
                                 const float* __restrict__ conv_b,
                                 float* __restrict__ xs_act,
                                 unsigned short* __restrict__ xs_bf) {
  int idx = blockIdx.x * blockDim.x + threadIdx.x;
  const int total = ROWS * D_INNER;
  if (idx >= total) return;
  int d  = idx & (D_INNER - 1);
  int bl = idx >> 10;            // D_INNER == 1024
  int l  = bl & (SEQLEN - 1);
  int b  = bl >> 11;             // SEQLEN == 2048

  float acc = conv_b[d];
#pragma unroll
  for (int k = 0; k < D_CONV; ++k) {
    int ls = l - (D_CONV - 1) + k;
    if (ls >= 0)
      acc += conv_w[d * D_CONV + k] *
             xz[((size_t)b * SEQLEN + ls) * (2 * D_INNER) + d];
  }
  float s = acc / (1.f + __expf(-acc));   // SiLU
  xs_act[idx] = s;
  xs_bf[idx]  = f32_to_bf16_rne(s);
}

// ---------------- split dt columns of x_dbl -> bf16 [ROWS, DT_RANK] --------
__global__ void dt_split_kernel(const float* __restrict__ x_dbl,
                                unsigned short* __restrict__ dt_bf) {
  int idx = blockIdx.x * blockDim.x + threadIdx.x;
  if (idx >= ROWS * DT_RANK) return;
  int r = idx >> 5;              // DT_RANK == 32
  int c = idx & (DT_RANK - 1);
  dt_bf[idx] = f32_to_bf16_rne(x_dbl[(size_t)r * (DT_RANK + 2 * D_STATE) + c]);
}

// ---------------- delta = softplus(gemm_out + dt_proj_b), in place ---------
__global__ void softplus_bias_kernel(float* __restrict__ delta,
                                     const float* __restrict__ bias) {
  int idx = blockIdx.x * blockDim.x + threadIdx.x;
  if (idx >= ROWS * D_INNER) return;
  int d = idx & (D_INNER - 1);
  float v = delta[idx] + bias[d];
  delta[idx] = (v > 20.f) ? v : __logf(1.f + __expf(v));
}

// ---------------- selective scan ----------------
// Thread t handles (b, d, n): p = t>>4 is the (b,d) channel, n = t&15.
__global__ void scan_kernel(const float* __restrict__ delta,
                            const float* __restrict__ xs_act,
                            const float* __restrict__ xz,
                            const float* __restrict__ x_dbl,
                            const float* __restrict__ A_log,
                            const float* __restrict__ D_param,
                            unsigned short* __restrict__ y_bf) {
  int t = blockIdx.x * blockDim.x + threadIdx.x;
  if (t >= BATCH * D_INNER * D_STATE) return;
  const int n = t & (D_STATE - 1);
  const int p = t >> 4;                 // channel index
  const int d = p & (D_INNER - 1);
  const int b = p >> 10;

  const float An = -__expf(A_log[d * D_STATE + n]);
  const float Dp = D_param[d];
  float h = 0.f;

  for (int l = 0; l < SEQLEN; ++l) {
    const size_t row = (size_t)b * SEQLEN + l;
    const float dlt = delta[row * D_INNER + d];
    const float u   = xs_act[row * D_INNER + d];
    const float* bc = x_dbl + row * (DT_RANK + 2 * D_STATE);
    const float Bn  = bc[DT_RANK + n];
    const float Cn  = bc[DT_RANK + D_STATE + n];

    const float dA = __expf(dlt * An);
    h = dA * h + (dlt * Bn) * u;
    float acc = h * Cn;
    // reduce across the 16-lane state group (stays within half-wave)
    acc += __shfl_xor(acc, 8);
    acc += __shfl_xor(acc, 4);
    acc += __shfl_xor(acc, 2);
    acc += __shfl_xor(acc, 1);

    if (n == 0) {
      float zz = xz[row * (2 * D_INNER) + D_INNER + d];
      float y = acc + u * Dp;
      y *= zz / (1.f + __expf(-zz));    // * silu(z)
      y_bf[row * D_INNER + d] = f32_to_bf16_rne(y);
    }
  }
}

// ---------------------------------------------------------------------------
extern "C" void kernel_launch(void* const* d_in, const int* in_sizes, int n_in,
                              void* d_out, int out_size, void* d_ws, size_t ws_size,
                              hipStream_t stream) {
  const float* x         = (const float*)d_in[0];
  const float* in_proj_w = (const float*)d_in[1];
  const float* conv_w    = (const float*)d_in[2];
  const float* conv_b    = (const float*)d_in[3];
  const float* x_proj_w  = (const float*)d_in[4];
  const float* dt_proj_w = (const float*)d_in[5];
  const float* dt_proj_b = (const float*)d_in[6];
  const float* A_log     = (const float*)d_in[7];
  const float* D_param   = (const float*)d_in[8];
  const float* out_proj_w= (const float*)d_in[9];
  float* out = (float*)d_out;

  char* ws = (char*)d_ws;
  size_t off = 0;
  auto alloc = [&](size_t bytes) -> char* {
    char* p = ws + off;
    off += (bytes + 255) & ~(size_t)255;
    return p;
  };

  unsigned short* x_bf   = (unsigned short*)alloc((size_t)ROWS * D_MODEL * 2);
  unsigned short* wi_bf  = (unsigned short*)alloc((size_t)2 * D_INNER * D_MODEL * 2);
  unsigned short* wx_bf  = (unsigned short*)alloc((size_t)(DT_RANK + 2 * D_STATE) * D_INNER * 2);
  unsigned short* wdt_bf = (unsigned short*)alloc((size_t)D_INNER * DT_RANK * 2);
  unsigned short* wo_bf  = (unsigned short*)alloc((size_t)D_MODEL * D_INNER * 2);
  float* xz              = (float*)alloc((size_t)ROWS * 2 * D_INNER * 4);
  float* xs_act          = (float*)alloc((size_t)ROWS * D_INNER * 4);
  unsigned short* xs_bf  = (unsigned short*)alloc((size_t)ROWS * D_INNER * 2);
  float* x_dbl           = (float*)alloc((size_t)ROWS * (DT_RANK + 2 * D_STATE) * 4);
  unsigned short* dt_bf  = (unsigned short*)alloc((size_t)ROWS * DT_RANK * 2);
  float* delta           = (float*)alloc((size_t)ROWS * D_INNER * 4);
  unsigned short* y_bf   = (unsigned short*)alloc((size_t)ROWS * D_INNER * 2);

  const int CT = 256;
  auto cvt = [&](const float* src, unsigned short* dst, int n) {
    cvt_bf16_kernel<<<(n + CT - 1) / CT, CT, 0, stream>>>(src, dst, n);
  };
  cvt(x,          x_bf,   ROWS * D_MODEL);
  cvt(in_proj_w,  wi_bf,  2 * D_INNER * D_MODEL);
  cvt(x_proj_w,   wx_bf,  (DT_RANK + 2 * D_STATE) * D_INNER);
  cvt(dt_proj_w,  wdt_bf, D_INNER * DT_RANK);
  cvt(out_proj_w, wo_bf,  D_MODEL * D_INNER);

  auto gemm = [&](const unsigned short* A, const unsigned short* W, float* C,
                  int M, int N, int K) {
    dim3 grid(N / 64, M / 128);
    wmma_gemm_bf16_kernel<<<grid, 128, 0, stream>>>(
        (const __bf16*)A, (const __bf16*)W, C, M, N, K);
  };

  // 1) in_proj: xz[4096,2048] = x * in_proj_w^T
  gemm(x_bf, wi_bf, xz, ROWS, 2 * D_INNER, D_MODEL);

  // 2) depthwise causal conv + SiLU
  {
    int n = ROWS * D_INNER;
    conv_silu_kernel<<<(n + CT - 1) / CT, CT, 0, stream>>>(
        xz, conv_w, conv_b, xs_act, xs_bf);
  }

  // 3) x_proj: x_dbl[4096,64] = xs * x_proj_w^T
  gemm(xs_bf, wx_bf, x_dbl, ROWS, DT_RANK + 2 * D_STATE, D_INNER);

  // 4) split dt columns to bf16
  {
    int n = ROWS * DT_RANK;
    dt_split_kernel<<<(n + CT - 1) / CT, CT, 0, stream>>>(x_dbl, dt_bf);
  }

  // 5) dt_proj (K=32: one WMMA step), then bias + softplus
  gemm(dt_bf, wdt_bf, delta, ROWS, D_INNER, DT_RANK);
  {
    int n = ROWS * D_INNER;
    softplus_bias_kernel<<<(n + CT - 1) / CT, CT, 0, stream>>>(delta, dt_proj_b);
  }

  // 6) selective scan (+ D skip + silu(z) gate), emits y in bf16
  {
    int n = BATCH * D_INNER * D_STATE;
    scan_kernel<<<(n + CT - 1) / CT, CT, 0, stream>>>(
        delta, xs_act, xz, x_dbl, A_log, D_param, y_bf);
  }

  // 7) out_proj: out[4096,512] = y * out_proj_w^T
  gemm(y_bf, wo_bf, out, ROWS, D_MODEL, D_INNER);
}
